// Qwen3Attention_23751169147199
// MI455X (gfx1250) — matrix operand order repaired and checked by reading, not compiled, not run on hardware
//
#include <hip/hip_runtime.h>
#include <stdint.h>

typedef __attribute__((ext_vector_type(16))) __bf16 v16bf;
typedef __attribute__((ext_vector_type(8)))  float  v8f;
typedef __attribute__((ext_vector_type(4)))  int    v4i;

union AFrag { v16bf v; uint4 q[2]; };

#define S_LEN 1024
#define DMODEL 4096
#define NH 32
#define KVH 8
#define HD 128

#if defined(__AMDGCN__) && defined(__has_builtin)
#if __has_builtin(__builtin_amdgcn_global_load_async_to_lds_b128)
#define USE_ASYNC_LDS 1
#endif
#endif

// copy 16 bytes global -> LDS (async on CDNA5 when available)
__device__ __forceinline__ void copy16_g2l(const __bf16* __restrict__ g, __bf16* l) {
#ifdef USE_ASYNC_LDS
    __builtin_amdgcn_global_load_async_to_lds_b128((v4i*)g, (v4i*)l, 0, 0);
#else
    *reinterpret_cast<uint4*>(l) = *reinterpret_cast<const uint4*>(g);
#endif
}

__device__ __forceinline__ void async_wait_all() {
#ifdef USE_ASYNC_LDS
#if __has_builtin(__builtin_amdgcn_s_wait_asynccnt)
    __builtin_amdgcn_s_wait_asynccnt(0);
#else
    asm volatile("s_wait_asynccnt 0" ::: "memory");
#endif
#endif
}

// ---------------- elementwise casts ----------------
__global__ void cast_bf16(const float* __restrict__ src, __bf16* __restrict__ dst, int n) {
    int i = blockIdx.x * blockDim.x + threadIdx.x;
    int stride = gridDim.x * blockDim.x;
    for (; i < n; i += stride) dst[i] = (__bf16)src[i];
}

// transpose cast: src [K,N] f32 -> dst [N,K] bf16
__global__ void tcast_bf16(const float* __restrict__ src, __bf16* __restrict__ dst,
                           int K, int N) {
    int i = blockIdx.x * blockDim.x + threadIdx.x;
    int total = K * N;
    int stride = gridDim.x * blockDim.x;
    for (; i < total; i += stride) {
        int k = i / N;
        int n = i - k * N;
        dst[(size_t)n * K + k] = (__bf16)src[i];
    }
}

// v: [B,S,KVH,HD] f32 -> vt: [B,KVH,HD,S] bf16
__global__ void vtrans_bf16(const float* __restrict__ v, __bf16* __restrict__ vt, int n) {
    int i = blockIdx.x * blockDim.x + threadIdx.x;
    if (i >= n) return;
    int hd = i & (HD - 1);
    int t  = i >> 7;
    int kh = t & (KVH - 1); t >>= 3;
    int s  = t & (S_LEN - 1); t >>= 10;
    int b  = t;
    vt[(((size_t)(b * KVH + kh)) * HD + hd) * S_LEN + s] = (__bf16)v[i];
}

// ---------------- RMSNorm + RoPE ----------------
__global__ __launch_bounds__(128) void normrope(const float* __restrict__ src,
                                                __bf16* __restrict__ dst,
                                                const float* __restrict__ w,
                                                const int* __restrict__ pos,
                                                int nheads) {
    __shared__ float red[128];
    __shared__ float nv[128];
    int h = threadIdx.x;
    int s = blockIdx.x, head = blockIdx.y, b = blockIdx.z;
    size_t base = (((size_t)(b * S_LEN + s)) * nheads + head) * HD;
    float t = src[base + h];
    red[h] = t * t;
    __syncthreads();
    for (int off = 64; off > 0; off >>= 1) {
        if (h < off) red[h] += red[h + off];
        __syncthreads();
    }
    float rs  = rsqrtf(red[0] * (1.0f / 128.0f) + 1e-6f);
    float nrm = t * rs * w[h];
    nv[h] = nrm;
    __syncthreads();
    int j = h & 63;
    float p    = (float)pos[b * S_LEN + s];
    float invf = __expf(-(float)j * (0.015625f * logf(1.0e6f)));
    float ang  = p * invf;
    float c, sn;
    __sincosf(ang, &sn, &c);
    float n1 = nv[j], n2 = nv[j + 64];
    float o  = (h < 64) ? (n1 * c - n2 * sn) : (n2 * c + n1 * sn);
    dst[base + h] = (__bf16)o;
}

// ---------------- tiled bf16 WMMA GEMM (B pre-transposed) ----------------
// C[M,N] f32 = A[M,K] * Bt[N,K]^T, all bf16 inputs. M%128==0, N%128==0, K%32==0.
#define BM 128
#define BN 128
#define BK 32

__global__ __launch_bounds__(256) void gemm_bf16_tn(const __bf16* __restrict__ A,
                                                    const __bf16* __restrict__ Bt,
                                                    float* __restrict__ C,
                                                    int M, int N, int K) {
    __shared__ __bf16 As[BM * BK];  // [m][k]
    __shared__ __bf16 Bs[BN * BK];  // [n][k]
    int tid  = threadIdx.x;
    int lane = tid & 31;
    int w    = tid >> 5;
    int wm   = w >> 1;  // 0..3 : wave M tile (32 rows)
    int wn   = w & 1;   // 0..1 : wave N tile (64 cols)
    int mBase = blockIdx.y * BM;
    int nBase = blockIdx.x * BN;
    int l15 = lane & 15;
    int hi  = lane >> 4;

    v8f acc[2][4];
#pragma unroll
    for (int i = 0; i < 2; ++i)
#pragma unroll
        for (int j = 0; j < 4; ++j) acc[i][j] = v8f{};

    for (int k0 = 0; k0 < K; k0 += BK) {
        __syncthreads();  // previous tile fully consumed
        // stage A tile: 128 rows x 32 k, 16B chunks (512 chunks, 2/thread)
#pragma unroll
        for (int i = 0; i < 2; ++i) {
            int idx = tid + i * 256;
            int r = idx >> 2;
            int c = (idx & 3) * 8;
            copy16_g2l(A + (size_t)(mBase + r) * K + k0 + c, &As[r * BK + c]);
        }
        // stage B tile from Bt rows: 128 n x 32 k
#pragma unroll
        for (int i = 0; i < 2; ++i) {
            int idx = tid + i * 256;
            int r = idx >> 2;
            int c = (idx & 3) * 8;
            copy16_g2l(Bt + (size_t)(nBase + r) * K + k0 + c, &Bs[r * BK + c]);
        }
        async_wait_all();
        __syncthreads();

        // A fragments (reused across 4 N tiles)
        AFrag a[2];
        int kbA = hi ? 8 : 0;
#pragma unroll
        for (int mt = 0; mt < 2; ++mt) {
            int m = wm * 32 + mt * 16 + l15;
            a[mt].q[0] = *reinterpret_cast<const uint4*>(&As[m * BK + kbA]);
            a[mt].q[1] = *reinterpret_cast<const uint4*>(&As[m * BK + 16 + kbA]);
        }
        int kbB = hi ? 8 : 0;
#pragma unroll
        for (int nt = 0; nt < 4; ++nt) {
            int n = wn * 64 + nt * 16 + l15;
            AFrag bf;
            bf.q[0] = *reinterpret_cast<const uint4*>(&Bs[n * BK + kbB]);
            bf.q[1] = *reinterpret_cast<const uint4*>(&Bs[n * BK + kbB + 16]);
#pragma unroll
            for (int mt = 0; mt < 2; ++mt) {
                acc[mt][nt] = __builtin_amdgcn_wmma_f32_16x16x32_bf16(
                    false, a[mt].v, false, bf.v, (short)0, acc[mt][nt], false, false);
            }
        }
    }

    size_t cbase = (size_t)(mBase + wm * 32 + 8 * hi) * N + nBase + wn * 64 + l15;
#pragma unroll
    for (int mt = 0; mt < 2; ++mt)
#pragma unroll
        for (int r = 0; r < 8; ++r) {
            size_t rowoff = cbase + (size_t)(mt * 16 + r) * N;
#pragma unroll
            for (int nt = 0; nt < 4; ++nt)
                C[rowoff + nt * 16] = acc[mt][nt][r];
        }
}

// ---------------- single-wave flash attention ----------------
// Q:  [B,S,NH,HD]  bf16 (post norm+rope)
// Kc: [B,S,KVH,HD] bf16 (post norm+rope)
// Vt: [B,KVH,HD,S] bf16
// O:  [B,S,NH,HD]  bf16
__global__ __launch_bounds__(32) void attn_kernel(const __bf16* __restrict__ Q,
                                                  const __bf16* __restrict__ Kc,
                                                  const __bf16* __restrict__ Vt,
                                                  __bf16* __restrict__ O) {
    __shared__ __bf16 Pld[16 * 32];
    int lane = threadIdx.x;
    int l15 = lane & 15, hi = lane >> 4;
    int qt = blockIdx.x;
    int head = blockIdx.y;
    int b = blockIdx.z;
    int kvh = head >> 2;  // GQA groups = 4
    int qBase = qt * 16;
    const float sm = 0.08838834764831845f;  // 128^-0.5

    AFrag qf[4];
    {
        const __bf16* qrow = Q + (((size_t)(b * S_LEN + qBase + l15)) * NH + head) * HD;
        int kb = hi ? 8 : 0;
#pragma unroll
        for (int c = 0; c < 4; ++c) {
            qf[c].q[0] = *reinterpret_cast<const uint4*>(qrow + c * 32 + kb);
            qf[c].q[1] = *reinterpret_cast<const uint4*>(qrow + c * 32 + 16 + kb);
        }
    }

    v8f o[8];
#pragma unroll
    for (int i = 0; i < 8; ++i) o[i] = v8f{};
    float mrow[8], lrow[8];
#pragma unroll
    for (int r = 0; r < 8; ++r) { mrow[r] = -3.0e38f; lrow[r] = 0.0f; }

    int qEnd = qBase + 15;
    for (int kb0 = 0; kb0 <= qEnd; kb0 += 32) {
        v8f s[2];
        s[0] = v8f{}; s[1] = v8f{};
#pragma unroll
        for (int half = 0; half < 2; ++half) {
            int t = kb0 + half * 16 + l15;
            const __bf16* krow = Kc + (((size_t)(b * S_LEN + t)) * KVH + kvh) * HD;
            int kb = hi ? 8 : 0;
#pragma unroll
            for (int c = 0; c < 4; ++c) {
                AFrag kf;
                kf.q[0] = *reinterpret_cast<const uint4*>(krow + c * 32 + kb);
                kf.q[1] = *reinterpret_cast<const uint4*>(krow + c * 32 + kb + 16);
                s[half] = __builtin_amdgcn_wmma_f32_16x16x32_bf16(
                    false, qf[c].v, false, kf.v, (short)0, s[half], false, false);
            }
        }
        __syncthreads();
#pragma unroll
        for (int r = 0; r < 8; ++r) {
            int m = r + 8 * hi;
            int qpos = qBase + m;
            float a0 = s[0][r] * sm; if (kb0 + l15 > qpos)      a0 = -3.0e38f;
            float a1 = s[1][r] * sm; if (kb0 + 16 + l15 > qpos) a1 = -3.0e38f;
            float mx = fmaxf(a0, a1);
#pragma unroll
            for (int off = 8; off > 0; off >>= 1) mx = fmaxf(mx, __shfl_xor(mx, off, 32));
            float mn   = fmaxf(mrow[r], mx);
            float corr = __expf(mrow[r] - mn);
            float p0 = __expf(a0 - mn), p1 = __expf(a1 - mn);
            float rs = p0 + p1;
#pragma unroll
            for (int off = 8; off > 0; off >>= 1) rs += __shfl_xor(rs, off, 32);
            lrow[r] = lrow[r] * corr + rs;
            mrow[r] = mn;
#pragma unroll
            for (int nc = 0; nc < 8; ++nc) o[nc][r] *= corr;
            Pld[m * 32 + l15]      = (__bf16)p0;
            Pld[m * 32 + 16 + l15] = (__bf16)p1;
        }
        __syncthreads();
        AFrag pf;
        {
            int kb = hi ? 8 : 0;
            pf.q[0] = *reinterpret_cast<const uint4*>(&Pld[l15 * 32 + kb]);
            pf.q[1] = *reinterpret_cast<const uint4*>(&Pld[l15 * 32 + 16 + kb]);
        }
        int kbv = hi ? 8 : 0;
#pragma unroll
        for (int nc = 0; nc < 8; ++nc) {
            const __bf16* vrow =
                Vt + (((size_t)(b * KVH + kvh)) * HD + nc * 16 + l15) * S_LEN + kb0 + kbv;
            AFrag vf;
            vf.q[0] = *reinterpret_cast<const uint4*>(vrow);
            vf.q[1] = *reinterpret_cast<const uint4*>(vrow + 16);
            o[nc] = __builtin_amdgcn_wmma_f32_16x16x32_bf16(
                false, pf.v, false, vf.v, (short)0, o[nc], false, false);
        }
    }

    float rl[8];
#pragma unroll
    for (int r = 0; r < 8; ++r) rl[r] = 1.0f / lrow[r];
#pragma unroll
    for (int nc = 0; nc < 8; ++nc)
#pragma unroll
        for (int r = 0; r < 8; ++r) {
            int m = r + 8 * hi;
            O[(((size_t)(b * S_LEN + qBase + m)) * NH + head) * HD + nc * 16 + l15] =
                (__bf16)(o[nc][r] * rl[r]);
        }
}

// ---------------- host launcher ----------------
extern "C" void kernel_launch(void* const* d_in, const int* in_sizes, int n_in,
                              void* d_out, int out_size, void* d_ws, size_t ws_size,
                              hipStream_t stream) {
    const float* x   = (const float*)d_in[0];
    const int*   pos = (const int*)d_in[1];
    const float* wq  = (const float*)d_in[2];
    const float* wk  = (const float*)d_in[3];
    const float* wv  = (const float*)d_in[4];
    const float* wo  = (const float*)d_in[5];
    const float* qnw = (const float*)d_in[6];
    const float* knw = (const float*)d_in[7];
    float* out = (float*)d_out;

    const int BS  = 2 * S_LEN;  // 2048
    const int NQ  = NH * HD;    // 4096
    const int NKV = KVH * HD;   // 1024

    char* ws = (char*)d_ws;
    size_t off = 0;
    auto alloc = [&](size_t bytes) {
        void* p = ws + off;
        off += (bytes + 255) & ~(size_t)255;
        return p;
    };
    __bf16* x_bf  = (__bf16*)alloc((size_t)BS * DMODEL * 2);
    __bf16* wqT   = (__bf16*)alloc((size_t)NQ * DMODEL * 2);   // [NQ, D]
    __bf16* wkT   = (__bf16*)alloc((size_t)NKV * DMODEL * 2);  // [NKV, D]
    __bf16* wvT   = (__bf16*)alloc((size_t)NKV * DMODEL * 2);
    __bf16* woT   = (__bf16*)alloc((size_t)DMODEL * NQ * 2);   // [D, NQ]
    float*  q_f32 = (float*)alloc((size_t)BS * NQ * 4);
    float*  k_f32 = (float*)alloc((size_t)BS * NKV * 4);
    float*  v_f32 = (float*)alloc((size_t)BS * NKV * 4);
    __bf16* q_bf  = (__bf16*)alloc((size_t)BS * NQ * 2);
    __bf16* k_bf  = (__bf16*)alloc((size_t)BS * NKV * 2);
    __bf16* vT_bf = (__bf16*)alloc((size_t)BS * NKV * 2);
    __bf16* at_bf = (__bf16*)alloc((size_t)BS * NQ * 2);

    // 1) casts (weights transposed to [N,K] so GEMM tiles load contiguously)
    cast_bf16<<<4096, 256, 0, stream>>>(x, x_bf, BS * DMODEL);
    tcast_bf16<<<4096, 256, 0, stream>>>(wq, wqT, DMODEL, NQ);
    tcast_bf16<<<4096, 256, 0, stream>>>(wk, wkT, DMODEL, NKV);
    tcast_bf16<<<4096, 256, 0, stream>>>(wv, wvT, DMODEL, NKV);
    tcast_bf16<<<4096, 256, 0, stream>>>(wo, woT, NQ, DMODEL);

    // 2) projections
    gemm_bf16_tn<<<dim3(NQ / BN, BS / BM), 256, 0, stream>>>(x_bf, wqT, q_f32, BS, NQ, DMODEL);
    gemm_bf16_tn<<<dim3(NKV / BN, BS / BM), 256, 0, stream>>>(x_bf, wkT, k_f32, BS, NKV, DMODEL);
    gemm_bf16_tn<<<dim3(NKV / BN, BS / BM), 256, 0, stream>>>(x_bf, wvT, v_f32, BS, NKV, DMODEL);

    // 3) RMSNorm + RoPE -> bf16 ; V transpose-cast
    normrope<<<dim3(S_LEN, NH, 2), 128, 0, stream>>>(q_f32, q_bf, qnw, pos, NH);
    normrope<<<dim3(S_LEN, KVH, 2), 128, 0, stream>>>(k_f32, k_bf, knw, pos, KVH);
    {
        int n = BS * NKV;
        vtrans_bf16<<<(n + 255) / 256, 256, 0, stream>>>(v_f32, vT_bf, n);
    }

    // 4) causal GQA flash attention
    attn_kernel<<<dim3(S_LEN / 16, NH, 2), 32, 0, stream>>>(q_bf, k_bf, vT_bf, at_bf);

    // 5) output projection -> f32 d_out
    gemm_bf16_tn<<<dim3(DMODEL / BN, BS / BM), 256, 0, stream>>>(at_bf, woT, out, BS, DMODEL, NQ);

    (void)in_sizes; (void)n_in; (void)out_size; (void)ws_size;
}
